// ScaledDotProductAttention_2035814499006
// MI455X (gfx1250) — matrix operand order repaired and checked by reading, not compiled, not run on hardware
//
#include <hip/hip_runtime.h>

typedef __bf16 v16bf __attribute__((ext_vector_type(16)));
typedef __bf16 v8bf  __attribute__((ext_vector_type(8)));
typedef __bf16 v4bf  __attribute__((ext_vector_type(4)));
typedef float  v8f   __attribute__((ext_vector_type(8)));

#define B_    32
#define L_    1024
#define D_    128
#define SROW  1028              // padded S row stride (floats)
#define WAVES 4
#define KSTR  136               // bf16 units per K-tile row (128 + 8 pad)
#define VSTR  40                // bf16 units per V^T row   (32 + 8 pad)

#define STATS_OFF (WAVES * 16 * SROW)          // floats
#define KBUF_OFF  (STATS_OFF + WAVES * 32)     // floats
#define KBUF_BF   (2 * 16 * KSTR)              // bf16 units, both K buffers
#define VBUF_BF   (2 * 128 * VSTR)             // bf16 units, both V buffers
#define SMEM_BYTES ((size_t)KBUF_OFF * 4 + (size_t)KBUF_BF * 2 + (size_t)VBUF_BF * 2)

__device__ __forceinline__ __bf16 f2bf(float f) { return (__bf16)f; }

// A-matrix (16x32 bf16) per-lane element->K map (contiguous, matches B):
// element e: j=e/2, p=e&1, K = half*16 + 2*j + p
__device__ __forceinline__ int a_kk0(int j, int half) {
  return half * 16 + (j << 1);
}

// 32-byte (16 x bf16) fragment load from LDS as two b128s
__device__ __forceinline__ v16bf ld_frag32(const __bf16* p) {
  const v8bf* pv = (const v8bf*)p;
  v8bf lo = pv[0], hi = pv[1];
  return __builtin_shufflevector(lo, hi, 0, 1, 2, 3, 4, 5, 6, 7,
                                 8, 9, 10, 11, 12, 13, 14, 15);
}

__global__ void attn_fused_kernel(const float* __restrict__ q,
                                  const float* __restrict__ k,
                                  const float* __restrict__ v,
                                  float* __restrict__ out_f,
                                  float* __restrict__ attn_f) {
  extern __shared__ float smem[];
  const int tid  = threadIdx.x;
  const int wave = tid >> 5;
  const int lane = tid & 31;
  const int m16  = lane & 15;
  const int half = lane >> 4;

  const int b     = blockIdx.x >> 4;     // 16 query-blocks per batch
  const int qblk  = blockIdx.x & 15;
  const int qbase = qblk * 64 + wave * 16;

  float*  Sw    = smem + (size_t)wave * 16 * SROW;
  float*  stats = smem + STATS_OFF + wave * 32;
  __bf16* Kbuf  = (__bf16*)(smem + KBUF_OFF);
  __bf16* Vbuf  = Kbuf + KBUF_BF;

  const size_t base = (size_t)b * L_ * D_;

  // ---------------- Q A-fragments: 16x32 bf16, 4 chunks over D=128 ----------
  v16bf aq[4];
  #pragma unroll
  for (int c = 0; c < 4; ++c) {
    const float* qrow = q + base + (size_t)(qbase + m16) * D_ + c * 32;
    #pragma unroll
    for (int j = 0; j < 8; ++j) {
      float2 f = *(const float2*)(qrow + a_kk0(j, half));
      aq[c][2 * j]     = f2bf(f.x);
      aq[c][2 * j + 1] = f2bf(f.y);
    }
  }

  // ---------------- Phase 1: S = Q K^T over 64 column tiles ------------------
  // K tiles staged cooperatively into LDS as bf16 (double-buffered).
  #pragma unroll 1
  for (int nt = 0; nt < 64; ++nt) {
    const int n0 = nt * 16;
    __bf16* kt_buf = Kbuf + (nt & 1) * (16 * KSTR);
    {   // stage: 128 threads, 16 rows x 128 cols, 16 f32 each -> bf16
      const int r  = tid >> 3;
      const int cb = (tid & 7) * 16;
      const float* src = k + base + (size_t)(n0 + r) * D_ + cb;
      v4bf* dst = (v4bf*)(kt_buf + r * KSTR + cb);
      #pragma unroll
      for (int t = 0; t < 4; ++t) {
        float4 f = *(const float4*)(src + 4 * t);
        v4bf h;
        h[0] = f2bf(f.x); h[1] = f2bf(f.y); h[2] = f2bf(f.z); h[3] = f2bf(f.w);
        dst[t] = h;
      }
    }
    __syncthreads();

    // B layout: lane = N (K-row), element e -> K = half*16 + e (contiguous)
    v16bf bk0 = ld_frag32(kt_buf + m16 * KSTR + 0 * 32 + half * 16);
    v16bf bk1 = ld_frag32(kt_buf + m16 * KSTR + 1 * 32 + half * 16);
    v16bf bk2 = ld_frag32(kt_buf + m16 * KSTR + 2 * 32 + half * 16);
    v16bf bk3 = ld_frag32(kt_buf + m16 * KSTR + 3 * 32 + half * 16);

    v8f acc0 = {}, acc1 = {};
    acc0 = __builtin_amdgcn_wmma_f32_16x16x32_bf16(false, aq[0], false, bk0,
                                                   (short)0, acc0, false, false);
    acc1 = __builtin_amdgcn_wmma_f32_16x16x32_bf16(false, aq[1], false, bk1,
                                                   (short)0, acc1, false, false);
    acc0 = __builtin_amdgcn_wmma_f32_16x16x32_bf16(false, aq[2], false, bk2,
                                                   (short)0, acc0, false, false);
    acc1 = __builtin_amdgcn_wmma_f32_16x16x32_bf16(false, aq[3], false, bk3,
                                                   (short)0, acc1, false, false);
    #pragma unroll
    for (int r = 0; r < 8; ++r) {
      const int m = r + 8 * half;                 // C/D layout: VGPR r -> row
      Sw[m * SROW + n0 + m16] = acc0[r] + acc1[r];
    }
  }
  __syncthreads();

  // ---------------- Softmax row stats (2 lanes per row, float4 scans) --------
  {
    const float4* row4 = (const float4*)(Sw + m16 * SROW + half * 512);
    float mx = -3.4e38f;
    #pragma unroll 2
    for (int i = 0; i < 128; ++i) {
      float4 f = row4[i];
      mx = fmaxf(mx, fmaxf(fmaxf(f.x, f.y), fmaxf(f.z, f.w)));
    }
    mx = fmaxf(mx, __shfl_xor(mx, 16, 32));
    float s = 0.0f;
    #pragma unroll 2
    for (int i = 0; i < 128; ++i) {
      float4 f = row4[i];
      s += __expf(f.x - mx) + __expf(f.y - mx) +
           __expf(f.z - mx) + __expf(f.w - mx);
    }
    s += __shfl_xor(s, 16, 32);
    if (half == 0) {
      stats[m16]      = mx;
      stats[16 + m16] = 1.0f / s;
    }
  }
  __syncthreads();

  // ------- Write P (float4) coalesced to global attn, overwrite S in LDS -----
  {
    float* attn_base = attn_f + ((size_t)b * L_ + qbase) * L_;
    #pragma unroll 1
    for (int m = 0; m < 16; ++m) {
      const float mx  = stats[m];
      const float inv = stats[16 + m];
      float4* srow4 = (float4*)(Sw + m * SROW);
      float4* grow4 = (float4*)(attn_base + (size_t)m * L_);
      #pragma unroll 1
      for (int c4 = lane; c4 < 256; c4 += 32) {
        float4 f = srow4[c4];
        float4 p;
        p.x = __expf(f.x - mx) * inv;
        p.y = __expf(f.y - mx) * inv;
        p.z = __expf(f.z - mx) * inv;
        p.w = __expf(f.w - mx) * inv;
        srow4[c4] = p;        // keep for P·V
        grow4[c4] = p;        // 134 MB output stream, 512B/instr coalesced
      }
    }
  }
  __syncthreads();

  // ---------------- Phase 2: O = P V, fused with *Q --------------------------
  // V tiles staged cooperatively into LDS as transposed bf16 (double-buffered).
  v8f zero = {};
  v8f oacc[8];
  #pragma unroll
  for (int dt = 0; dt < 8; ++dt) oacc[dt] = zero;

  #pragma unroll 1
  for (int kt = 0; kt < 32; ++kt) {
    const int kb = kt * 32;
    __bf16* vt_buf = Vbuf + (kt & 1) * (128 * VSTR);
    {   // stage transpose: VT[d][kk] = V[kb+kk][d], bf16
      const int r  = tid >> 2;             // kk row 0..31
      const int cb = (tid & 3) * 32;       // d block
      const float* src = v + base + (size_t)(kb + r) * D_ + cb;
      #pragma unroll
      for (int i = 0; i < 8; ++i) {
        float4 f = *(const float4*)(src + 4 * i);
        vt_buf[(cb + 4 * i + 0) * VSTR + r] = f2bf(f.x);
        vt_buf[(cb + 4 * i + 1) * VSTR + r] = f2bf(f.y);
        vt_buf[(cb + 4 * i + 2) * VSTR + r] = f2bf(f.z);
        vt_buf[(cb + 4 * i + 3) * VSTR + r] = f2bf(f.w);
      }
    }
    __syncthreads();

    // P A-fragment gathered from LDS (A pair map), converted to bf16
    v16bf ap;
    const float* prow = Sw + m16 * SROW + kb;
    #pragma unroll
    for (int j = 0; j < 8; ++j) {
      float2 f = *(const float2*)(prow + a_kk0(j, half));
      ap[2 * j]     = f2bf(f.x);
      ap[2 * j + 1] = f2bf(f.y);
    }
    #pragma unroll
    for (int dt = 0; dt < 8; ++dt) {
      const int d0 = dt * 16;
      // B[kk,n] = V[kb+kk, d0+n]; lane = n, element e -> kk = half*16+e
      v16bf bv = ld_frag32(vt_buf + (d0 + m16) * VSTR + half * 16);
      oacc[dt] = __builtin_amdgcn_wmma_f32_16x16x32_bf16(false, ap, false, bv,
                                                         (short)0, oacc[dt],
                                                         false, false);
    }
  }

  #pragma unroll
  for (int dt = 0; dt < 8; ++dt) {
    const int d0 = dt * 16;
    #pragma unroll
    for (int r = 0; r < 8; ++r) {
      const int m = r + 8 * half;
      const size_t idx = base + (size_t)(qbase + m) * D_ + d0 + m16;
      out_f[idx] = oacc[dt][r] * q[idx];
    }
  }
}

extern "C" void kernel_launch(void* const* d_in, const int* in_sizes, int n_in,
                              void* d_out, int out_size, void* d_ws, size_t ws_size,
                              hipStream_t stream) {
  (void)in_sizes; (void)n_in; (void)out_size; (void)d_ws; (void)ws_size;
  const float* q = (const float*)d_in[0];
  const float* k = (const float*)d_in[1];
  const float* v = (const float*)d_in[2];
  float* out_f  = (float*)d_out;
  float* attn_f = out_f + (size_t)B_ * L_ * D_;

  hipFuncSetAttribute((const void*)attn_fused_kernel,
                      hipFuncAttributeMaxDynamicSharedMemorySize, (int)SMEM_BYTES);

  dim3 grid(B_ * (L_ / 64));   // 512 blocks: one (batch, 64-row) slab each
  dim3 block(128);             // 4 waves -> 4 SIMD32s of one WGP
  attn_fused_kernel<<<grid, block, SMEM_BYTES, stream>>>(q, k, v, out_f, attn_f);
}
